// LocatorReaderConditioner_84439057039407
// MI455X (gfx1250) — matrix operand order repaired and checked by reading, not compiled, not run on hardware
//
#include <hip/hip_runtime.h>
#include <hip/hip_bf16.h>

// ---------------- shapes (fixed by the reference) ----------------
#define BATCH  4
#define QLEN   512
#define KLEN   4096
#define DMODEL 1024
#define NHEAD  16
#define DHEAD  64
#define SM_SCALE 0.125f   // dh^-0.5 = 64^-0.5

typedef __bf16 bf16_t;
typedef __attribute__((ext_vector_type(16))) __bf16 bf16x16;
typedef __attribute__((ext_vector_type(8)))  __bf16 bf16x8;
typedef __attribute__((ext_vector_type(4)))  __bf16 bf16x4;
typedef __attribute__((ext_vector_type(8)))  float  f32x8;
typedef __attribute__((ext_vector_type(4)))  unsigned int u32x4;
typedef __attribute__((ext_vector_type(8)))  unsigned int u32x8;

#define WMMA_BF16(a, b, c) \
  __builtin_amdgcn_wmma_f32_16x16x32_bf16(false, (a), false, (b), (short)0, (c), false, false)

union Frag16 { bf16x16 v; bf16x8 h[2]; };

// Load a 16-element bf16 fragment as two contiguous 8-element (16B) chunks.
static __device__ inline bf16x16 ldfrag(const bf16_t* p0, const bf16_t* p1) {
  Frag16 f;
  f.h[0] = *(const bf16x8*)p0;
  f.h[1] = *(const bf16x8*)p1;
  return f.v;
}

// ---------------- fp32 -> bf16 convert ----------------
__global__ void __launch_bounds__(256)
cvt_f32_bf16(const float* __restrict__ src, bf16_t* __restrict__ dst, int n4) {
  int i = blockIdx.x * blockDim.x + threadIdx.x;
  int stride = gridDim.x * blockDim.x;
  for (; i < n4; i += stride) {
    float4 f = ((const float4*)src)[i];
    bf16x4 o;
    o.x = (bf16_t)f.x; o.y = (bf16_t)f.y; o.z = (bf16_t)f.z; o.w = (bf16_t)f.w;
    *(bf16x4*)(dst + (size_t)i * 4) = o;
  }
}

// ---------------- tiled WMMA GEMM: C[M,N] = A[M,Kd] * W[Kd,N] ----------------
// 256 threads = 8 waves (4 M x 2 N); block tile 128x128; wave tile 32x64.
// Weight k-tile staged transposed in LDS so B-fragments are contiguous b128 loads.
// mode 0: store bf16 row-major [M,N]
// mode 1: store f32  row-major [M,N]
// mode 2: store bf16 transposed-per-head: Vt[b][h][d][kpos]  (M = BATCH*KLEN, N = DMODEL)
__global__ void __launch_bounds__(256)
gemm_bf16(const bf16_t* __restrict__ A, const bf16_t* __restrict__ W,
          void* __restrict__ Cout, int M, int Kd, int N, int mode)
{
  __shared__ __align__(16) bf16_t wt[128 * 32];  // [n_local][k_local], 8 KB

  const int tid    = threadIdx.x;
  const int lane   = tid & 31;
  const int wave   = tid >> 5;        // 0..7
  const int wave_m = wave >> 1;       // 0..3
  const int wave_n = wave & 1;        // 0..1
  const int r      = lane & 15;
  const int hi     = lane >> 4;

  const int nblk = blockIdx.x * 128;
  const int n0   = nblk + wave_n * 64;
  const int m0   = blockIdx.y * 128 + wave_m * 32;

  f32x8 acc[2][4];
  #pragma unroll
  for (int u = 0; u < 2; ++u)
    #pragma unroll
    for (int t = 0; t < 4; ++t)
      #pragma unroll
      for (int j = 0; j < 8; ++j) acc[u][t][j] = 0.0f;

  const bf16_t* arow0 = A + (size_t)(m0 + r) * Kd;
  const bf16_t* arow1 = A + (size_t)(m0 + 16 + r) * Kd;

  for (int k0 = 0; k0 < Kd; k0 += 32) {
    // ---- cooperative staging of W[k0:k0+32, nblk:nblk+128] transposed into LDS ----
    {
      const int kr = tid >> 4;            // 0..15
      const int nc = (tid & 15) * 8;      // 0,8,..,120
      #pragma unroll
      for (int s = 0; s < 2; ++s) {
        const int kk = kr + s * 16;
        bf16x8 w8 = *(const bf16x8*)(W + (size_t)(k0 + kk) * N + nblk + nc);
        #pragma unroll
        for (int e = 0; e < 8; ++e) wt[(nc + e) * 32 + kk] = w8[e];
      }
      if (k0 + 32 < Kd) // prefetch next weight k-tile (global_prefetch_b8)
        __builtin_prefetch(W + (size_t)(k0 + 32 + kr) * N + nblk + nc, 0, 0);
    }
    __syncthreads();

    // A fragments: 16x32 each, per-lane contiguous runs at k0+hi*8 and k0+16+hi*8
    bf16x16 af0 = ldfrag(arow0 + k0 + hi * 8, arow0 + k0 + 16 + hi * 8);
    bf16x16 af1 = ldfrag(arow1 + k0 + hi * 8, arow1 + k0 + 16 + hi * 8);

    // Preload all four B-fragments so the 8 WMMAs issue back-to-back
    // instead of stalling on a ds_load/wait per pair.
    bf16x16 bfrag[4];
    #pragma unroll
    for (int t = 0; t < 4; ++t) {
      const bf16_t* bp = &wt[(wave_n * 64 + t * 16 + r) * 32 + hi * 8];
      bfrag[t] = ldfrag(bp, bp + 16);
    }
    #pragma unroll
    for (int t = 0; t < 4; ++t) {
      acc[0][t] = WMMA_BF16(af0, bfrag[t], acc[0][t]);
      acc[1][t] = WMMA_BF16(af1, bfrag[t], acc[1][t]);
    }
    __syncthreads();
  }

  // ---- store: C row = m0 + 16u + j + 8*hi, col = n0 + 16*t + r ----
  #pragma unroll
  for (int u = 0; u < 2; ++u)
    #pragma unroll
    for (int t = 0; t < 4; ++t)
      #pragma unroll
      for (int j = 0; j < 8; ++j) {
        const int row = m0 + u * 16 + j + hi * 8;
        const int col = n0 + t * 16 + r;
        const float v = acc[u][t][j];
        if (mode == 0) {
          ((bf16_t*)Cout)[(size_t)row * N + col] = (bf16_t)v;
        } else if (mode == 1) {
          ((float*)Cout)[(size_t)row * N + col] = v;
        } else {
          const int b    = row >> 12;        // /KLEN
          const int kpos = row & (KLEN - 1);
          const int h    = col >> 6;         // /DHEAD
          const int d    = col & (DHEAD - 1);
          ((bf16_t*)Cout)[(((size_t)(b * NHEAD + h)) * DHEAD + d) * KLEN + kpos] = (bf16_t)v;
        }
      }
}

// Issue a TDM tensor_load_to_lds from wave-uniform descriptor groups (ISA 8.3/8.4).
static __device__ inline void tdm_load_2d(unsigned lds_addr, unsigned long long gaddr,
                                          u32x8 g1) {
  u32x4 g0;
  g0[0] = 1u;                                   // count=1, user descriptor
  g0[1] = lds_addr;                             // LDS byte address
  g0[2] = (unsigned)gaddr;                      // global_addr[31:0]
  g0[3] = (unsigned)((gaddr >> 32) & 0x1FFFFFFu) | (2u << 30);  // addr[56:32] | type=2
  asm volatile("tensor_load_to_lds %0, %1" :: "s"(g0), "s"(g1) : "memory");
}

// ---------------- flash attention ----------------
// grid (QLEN/64, NHEAD, BATCH); 128 threads = 4 waves; each wave owns a 16-row Q tile.
// K/V step tiles are shared by all 4 waves -> one TDM DMA each into LDS per step.
// Qp: (B*Q, D) bf16 row-major; Kp: (B*K, D) bf16 row-major
// Vt: (B, H, DHEAD, KLEN) bf16; bias: (B, K) f32; Obf: (B*Q, D) bf16
__global__ void __launch_bounds__(128)
flash_attn(const bf16_t* __restrict__ Qp, const bf16_t* __restrict__ Kp,
           const bf16_t* __restrict__ Vt, const float* __restrict__ bias,
           bf16_t* __restrict__ Obf)
{
  __shared__ __align__(16) bf16_t ktile[32 * 64];   // [kpos_local][d]
  __shared__ __align__(16) bf16_t vtile[64 * 32];   // [d][kpos_local]
  __shared__ __align__(16) bf16_t pbuf[4][16 * 32]; // wave-private P tiles

  const int tid  = threadIdx.x;
  const int lane = tid & 31;
  const int wave = tid >> 5;
  const int r    = lane & 15;
  const int hi   = lane >> 4;

  const int b  = blockIdx.z;
  const int h  = blockIdx.y;
  const int q0 = blockIdx.x * 64 + wave * 16;

  // Q fragments (two dh chunks of 32), loaded once
  const bf16_t* qrow = Qp + (size_t)(b * QLEN + q0 + r) * DMODEL + h * DHEAD;
  const bf16x16 qa0 = ldfrag(qrow +      hi * 8, qrow + 16 + hi * 8);
  const bf16x16 qa1 = ldfrag(qrow + 32 + hi * 8, qrow + 48 + hi * 8);

  const bf16_t* kbase = Kp + (size_t)b * KLEN * DMODEL + h * DHEAD;
  const bf16_t* vbase = Vt + (size_t)(b * NHEAD + h) * DHEAD * KLEN;
  const float*  brow  = bias + (size_t)b * KLEN;
  bf16_t* pb = pbuf[wave];

  // Static TDM descriptor group1 words (data_size=2B, workgroup_mask=0).
  // K tile: 32 rows x 64 cols, row stride DMODEL.
  u32x8 kg1;
  kg1[0] = 0x00010000u;                // data_size=1 (2 bytes)
  kg1[1] = ((unsigned)DHEAD) << 16;    // tensor_dim0 lo16 = 64
  kg1[2] = ((unsigned)KLEN) << 16;     // tensor_dim0 hi = 0 | tensor_dim1 lo16
  kg1[3] = ((unsigned)DHEAD) << 16;    // tensor_dim1 hi = 0 | tile_dim0 = 64
  kg1[4] = 32u;                        // tile_dim1 = 32, tile_dim2 = 0
  kg1[5] = (unsigned)DMODEL;           // tensor_dim0_stride = 1024
  kg1[6] = 0u; kg1[7] = 0u;
  // V tile: 64 rows (d) x 32 cols (kpos), row stride KLEN.
  u32x8 vg1;
  vg1[0] = 0x00010000u;
  vg1[1] = ((unsigned)KLEN) << 16;     // tensor_dim0 lo16 = 4096
  vg1[2] = ((unsigned)DHEAD) << 16;    // tensor_dim1 lo16 = 64
  vg1[3] = 32u << 16;                  // tile_dim0 = 32
  vg1[4] = (unsigned)DHEAD;            // tile_dim1 = 64
  vg1[5] = (unsigned)KLEN;             // tensor_dim0_stride = 4096
  vg1[6] = 0u; vg1[7] = 0u;

  const unsigned lds_k = (unsigned)(size_t)&ktile[0];
  const unsigned lds_v = (unsigned)(size_t)&vtile[0];

  f32x8 acc[4];
  float mrow[8], lrow[8];
  #pragma unroll
  for (int j = 0; j < 8; ++j) { mrow[j] = -1e30f; lrow[j] = 0.0f; }
  #pragma unroll
  for (int t = 0; t < 4; ++t)
    #pragma unroll
    for (int j = 0; j < 8; ++j) acc[t][j] = 0.0f;

  for (int kk = 0; kk < KLEN; kk += 32) {
    __syncthreads();   // protect K/V tiles from previous iteration's readers
    if (wave == 0) {
      tdm_load_2d(lds_k, (unsigned long long)(size_t)(kbase + (size_t)kk * DMODEL), kg1);
      tdm_load_2d(lds_v, (unsigned long long)(size_t)(vbase + kk), vg1);
      __builtin_amdgcn_s_wait_tensorcnt(0);
    }
    __syncthreads();   // K/V tiles ready

    // ---- S = Q * K^T for two 16-wide column subtiles (K read from LDS) ----
    f32x8 s0, s1;
    #pragma unroll
    for (int j = 0; j < 8; ++j) { s0[j] = 0.0f; s1[j] = 0.0f; }
    {
      const bf16_t* kr0 = ktile + (size_t)r * DHEAD;
      const bf16_t* kr1 = ktile + (size_t)(16 + r) * DHEAD;
      bf16x16 kb00 = ldfrag(kr0 +      hi * 8, kr0 + 16 + hi * 8);
      bf16x16 kb01 = ldfrag(kr0 + 32 + hi * 8, kr0 + 48 + hi * 8);
      bf16x16 kb10 = ldfrag(kr1 +      hi * 8, kr1 + 16 + hi * 8);
      bf16x16 kb11 = ldfrag(kr1 + 32 + hi * 8, kr1 + 48 + hi * 8);
      s0 = WMMA_BF16(qa0, kb00, s0);
      s0 = WMMA_BF16(qa1, kb01, s0);
      s1 = WMMA_BF16(qa0, kb10, s1);
      s1 = WMMA_BF16(qa1, kb11, s1);
    }

    const float bias0 = brow[kk + r];
    const float bias1 = brow[kk + 16 + r];

    float sv0[8], sv1[8], ms[8];
    #pragma unroll
    for (int j = 0; j < 8; ++j) {
      sv0[j] = s0[j] * SM_SCALE + bias0;
      sv1[j] = s1[j] * SM_SCALE + bias1;
      ms[j]  = fmaxf(sv0[j], sv1[j]);
    }
    #pragma unroll
    for (int d = 1; d < 16; d <<= 1)
      #pragma unroll
      for (int j = 0; j < 8; ++j) ms[j] = fmaxf(ms[j], __shfl_xor(ms[j], d, 32));

    float alpha[8], rs[8];
    #pragma unroll
    for (int j = 0; j < 8; ++j) {
      const float mn = fmaxf(mrow[j], ms[j]);
      alpha[j] = __expf(mrow[j] - mn);
      sv0[j]   = __expf(sv0[j] - mn);
      sv1[j]   = __expf(sv1[j] - mn);
      rs[j]    = sv0[j] + sv1[j];
      mrow[j]  = mn;
    }
    #pragma unroll
    for (int d = 1; d < 16; d <<= 1)
      #pragma unroll
      for (int j = 0; j < 8; ++j) rs[j] += __shfl_xor(rs[j], d, 32);
    #pragma unroll
    for (int j = 0; j < 8; ++j) lrow[j] = lrow[j] * alpha[j] + rs[j];
    #pragma unroll
    for (int t = 0; t < 4; ++t)
      #pragma unroll
      for (int j = 0; j < 8; ++j) acc[t][j] *= alpha[j];

    // ---- P (C-fragment layout) -> LDS -> A-fragment layout ----
    #pragma unroll
    for (int j = 0; j < 8; ++j) {
      const int row = j + hi * 8;
      pb[row * 32 + r]      = (bf16_t)sv0[j];
      pb[row * 32 + 16 + r] = (bf16_t)sv1[j];
    }
    asm volatile("s_wait_dscnt 0" ::: "memory");  // wave-internal LDS RAW fence
    const bf16_t* pr = pb + r * 32;
    bf16x16 pa = ldfrag(pr + hi * 8, pr + 16 + hi * 8);

    // ---- acc += P * V: preload all V fragments, then 4 back-to-back WMMAs ----
    bf16x16 vb[4];
    #pragma unroll
    for (int t = 0; t < 4; ++t) {
      const bf16_t* vp = vtile + (size_t)(t * 16 + r) * 32 + hi * 8;
      vb[t] = ldfrag(vp, vp + 16);
    }
    #pragma unroll
    for (int t = 0; t < 4; ++t)
      acc[t] = WMMA_BF16(pa, vb[t], acc[t]);
  }

  // ---- finalize: divide by l, store bf16 into (B*Q, D) for the output GEMM ----
  float inv[8];
  #pragma unroll
  for (int j = 0; j < 8; ++j) inv[j] = 1.0f / lrow[j];
  #pragma unroll
  for (int t = 0; t < 4; ++t)
    #pragma unroll
    for (int j = 0; j < 8; ++j) {
      const int row = b * QLEN + q0 + j + hi * 8;
      const int col = h * DHEAD + t * 16 + r;
      Obf[(size_t)row * DMODEL + col] = (bf16_t)(acc[t][j] * inv[j]);
    }
}

// ---------------- launch ----------------
extern "C" void kernel_launch(void* const* d_in, const int* in_sizes, int n_in,
                              void* d_out, int out_size, void* d_ws, size_t ws_size,
                              hipStream_t stream) {
  const float* query = (const float*)d_in[0];  // (B, Q, D)
  const float* memry = (const float*)d_in[1];  // (B, K, D)
  const float* bias  = (const float*)d_in[2];  // (B, K)
  const float* Wq    = (const float*)d_in[3];
  const float* Wk    = (const float*)d_in[4];
  const float* Wv    = (const float*)d_in[5];
  const float* Wo    = (const float*)d_in[6];

  const size_t nQ  = (size_t)BATCH * QLEN * DMODEL;   // 2,097,152
  const size_t nM  = (size_t)BATCH * KLEN * DMODEL;   // 16,777,216
  const size_t nW  = (size_t)DMODEL * DMODEL;         // 1,048,576

  // workspace layout (bf16 elements)
  char* ws = (char*)d_ws;
  size_t off = 0;
  auto alloc = [&](size_t elems) { bf16_t* p = (bf16_t*)(ws + off); off += elems * sizeof(bf16_t); return p; };
  bf16_t* qbf  = alloc(nQ);
  bf16_t* mbf  = alloc(nM);
  bf16_t* wqbf = alloc(nW);
  bf16_t* wkbf = alloc(nW);
  bf16_t* wvbf = alloc(nW);
  bf16_t* wobf = alloc(nW);
  bf16_t* Qp   = alloc(nQ);
  bf16_t* Kp   = alloc(nM);
  bf16_t* Vt   = alloc(nM);
  bf16_t* attn = alloc(nQ);
  if (off > ws_size) return;  // insufficient scratch

  auto cvt = [&](const float* s, bf16_t* d, size_t n) {
    int n4 = (int)(n / 4);
    int blocks = (n4 + 255) / 256; if (blocks > 4096) blocks = 4096;
    cvt_f32_bf16<<<blocks, 256, 0, stream>>>(s, d, n4);
  };
  cvt(query, qbf, nQ);
  cvt(memry, mbf, nM);
  cvt(Wq, wqbf, nW);
  cvt(Wk, wkbf, nW);
  cvt(Wv, wvbf, nW);
  cvt(Wo, wobf, nW);

  const int MQ = BATCH * QLEN;   // 2048
  const int MK = BATCH * KLEN;   // 16384

  gemm_bf16<<<dim3(DMODEL / 128, MQ / 128), 256, 0, stream>>>(qbf, wqbf, Qp, MQ, DMODEL, DMODEL, 0);
  gemm_bf16<<<dim3(DMODEL / 128, MK / 128), 256, 0, stream>>>(mbf, wkbf, Kp, MK, DMODEL, DMODEL, 0);
  gemm_bf16<<<dim3(DMODEL / 128, MK / 128), 256, 0, stream>>>(mbf, wvbf, Vt, MK, DMODEL, DMODEL, 2);

  flash_attn<<<dim3(QLEN / 64, NHEAD, BATCH), 128, 0, stream>>>(Qp, Kp, Vt, bias, attn);

  gemm_bf16<<<dim3(DMODEL / 128, MQ / 128), 256, 0, stream>>>(attn, wobf, (void*)d_out, MQ, DMODEL, DMODEL, 1);
}
